// _AttentionHead_61272003445052
// MI455X (gfx1250) — matrix operand order repaired and checked by reading, not compiled, not run on hardware
//
#include <hip/hip_runtime.h>
#include <hip/hip_bf16.h>
#include <math.h>

typedef __bf16 bf16;
typedef __attribute__((ext_vector_type(16))) __bf16 v16bf;
typedef __attribute__((ext_vector_type(8)))  float  v8f;

union AB16 {
    v16bf v;
    bf16  h[16];
    uint4 q[2];
};

#define S_LEN  4096
#define D_IN   1024
#define D_HEAD 128

// LDS byte address for gfx1250 LDS-operand instructions: low 32 bits of the
// flat address.
__device__ inline uint32_t lds_addr_u32(const void* p) {
    return (uint32_t)(uintptr_t)p;
}

// ---------------------------------------------------------------------------
// Projection: out[m,n] = (sum_k X[m,k]*W[k,n] + bias[n]) * out_scale -> bf16.
// 8 waves / block, wave owns 16 rows x 128 cols. W chunk staged transposed in
// LDS so B-matrix lanes read K 8h..8h+7 and 16+8h..16+8h+7 (CDNA5 B layout).
// transpose_out=1 writes V transposed: vt[b][n][s] (PV WMMA B layout).
// out_scale folds the softmax 1/sqrt(dk) into q.
// ---------------------------------------------------------------------------
__global__ __launch_bounds__(256)
void proj_kernel(const float* __restrict__ X, const float* __restrict__ W,
                 const float* __restrict__ bias, bf16* __restrict__ out,
                 int transpose_out, float out_scale) {
    __shared__ bf16 wt[D_HEAD * 32];  // wt[n*32 + kk] = W[kc+kk][n]

    const int tid    = threadIdx.x;
    const int wave   = tid >> 5;
    const int lane   = tid & 31;
    const int lane16 = lane & 15;
    const int hi     = lane >> 4;

    const int m0 = blockIdx.x * 128 + wave * 16;
    const float* xrow = X + (size_t)(m0 + lane16) * D_IN;
    const int kbA = hi * 8;

    const v8f zero8 = {0.f, 0.f, 0.f, 0.f, 0.f, 0.f, 0.f, 0.f};
    v8f acc[8];
#pragma unroll
    for (int t = 0; t < 8; ++t) acc[t] = zero8;

    for (int kc = 0; kc < D_IN; kc += 32) {
        __syncthreads();
#pragma unroll
        for (int i = 0; i < 16; ++i) {
            int idx = i * 256 + tid;       // 0..4095
            int kk  = idx >> 7;            // 0..31
            int nn  = idx & 127;           // 0..127
            wt[nn * 32 + kk] = (bf16)W[(size_t)(kc + kk) * D_HEAD + nn];
        }
        __syncthreads();

        AB16 a;
        {
            const float* p = xrow + kc + kbA;
#pragma unroll
            for (int t = 0; t < 8; ++t) a.h[t] = (bf16)p[t];
#pragma unroll
            for (int t = 0; t < 8; ++t) a.h[8 + t] = (bf16)p[16 + t];
        }

#pragma unroll
        for (int ns = 0; ns < 8; ++ns) {
            AB16 b;
            const bf16* wp = &wt[(ns * 16 + lane16) * 32 + hi * 8];
            b.q[0] = *(const uint4*)(wp);
            b.q[1] = *(const uint4*)(wp + 16);
            acc[ns] = __builtin_amdgcn_wmma_f32_16x16x32_bf16(
                false, a.v, false, b.v, (short)0, acc[ns], false, false);
        }
    }

#pragma unroll
    for (int ns = 0; ns < 8; ++ns) {
        const int n    = ns * 16 + lane16;
        const float bv_ = bias[n];
        if (!transpose_out) {
#pragma unroll
            for (int r = 0; r < 8; ++r) {
                int mm = m0 + r + hi * 8;
                out[(size_t)mm * D_HEAD + n] = (bf16)((acc[ns][r] + bv_) * out_scale);
            }
        } else {
#pragma unroll
            for (int r = 0; r < 8; ++r) {
                int mm = m0 + r + hi * 8;
                int bb = mm >> 12;
                int ss = mm & (S_LEN - 1);
                out[((size_t)bb * D_HEAD + n) * S_LEN + ss] =
                    (bf16)((acc[ns][r] + bv_) * out_scale);
            }
        }
    }
}

// ---------------------------------------------------------------------------
// Fused flash attention. Block = 8 waves = 128 query rows of one batch.
// K/V^T chunks (64 kv positions) are double-buffered in LDS via
// GLOBAL_LOAD_ASYNC_TO_LDS_B128 (ASYNCcnt), DMA of chunk j+1 overlapping the
// WMMA+softmax of chunk j. Each wave issues 8 async ops per chunk, so
// s_wait_asynccnt 8 after issuing the next chunk proves the current one done.
// ---------------------------------------------------------------------------
__global__ __launch_bounds__(256)
void attn_kernel(const bf16* __restrict__ qb, const bf16* __restrict__ kb,
                 const bf16* __restrict__ vt, float* __restrict__ out) {
    __shared__ bf16 kbuf[2][64 * D_HEAD];   // [kv 0..63][head 0..127]
    __shared__ bf16 vbuf[2][D_HEAD * 64];   // [head 0..127][kv 0..63]
    __shared__ bf16 lds_p[8 * 16 * 64];     // per-wave 16x64 P tile

    const int tid    = threadIdx.x;
    const int wave   = tid >> 5;
    const int lane   = tid & 31;
    const int lane16 = lane & 15;
    const int hi     = lane >> 4;
    const int b      = blockIdx.y;
    const int m0     = blockIdx.x * 128 + wave * 16;

    const bf16* qbase = qb + (size_t)b * S_LEN * D_HEAD;
    const bf16* kbase = kb + (size_t)b * S_LEN * D_HEAD;
    const bf16* vbase = vt + (size_t)b * D_HEAD * S_LEN;

    const uint32_t kbuf_lds[2] = {lds_addr_u32(&kbuf[0][0]), lds_addr_u32(&kbuf[1][0])};
    const uint32_t vbuf_lds[2] = {lds_addr_u32(&vbuf[0][0]), lds_addr_u32(&vbuf[1][0])};

    // ---- async stage of kv chunk jn into LDS buffer `sel` -----------------
    auto stage_chunk = [&](int sel, int jn) {
        // K: contiguous 16KB; one base VGPR, offset: covers both LDS & global.
        const uint32_t k_lds = kbuf_lds[sel] + (uint32_t)tid * 16u;
        const uint32_t k_vo  = (uint32_t)tid * 16u;
        const unsigned long long k_gb =
            (unsigned long long)(uintptr_t)(kbase + (size_t)jn * D_HEAD);
        asm volatile("global_load_async_to_lds_b128 %0, %1, %2 offset:0"
                     :: "v"(k_lds), "v"(k_vo), "s"(k_gb) : "memory");
        asm volatile("global_load_async_to_lds_b128 %0, %1, %2 offset:4096"
                     :: "v"(k_lds), "v"(k_vo), "s"(k_gb) : "memory");
        asm volatile("global_load_async_to_lds_b128 %0, %1, %2 offset:8192"
                     :: "v"(k_lds), "v"(k_vo), "s"(k_gb) : "memory");
        asm volatile("global_load_async_to_lds_b128 %0, %1, %2 offset:12288"
                     :: "v"(k_lds), "v"(k_vo), "s"(k_gb) : "memory");
        // V^T: 128 rows (stride 8192B in global) x 128B each.
        const unsigned long long v_gb =
            (unsigned long long)(uintptr_t)(vbase + jn);
#pragma unroll
        for (int r = 0; r < 4; ++r) {
            uint32_t c   = (uint32_t)tid + 256u * r;   // 16B-chunk index 0..1023
            uint32_t n   = c >> 3;                     // head row 0..127
            uint32_t co  = (c & 7u) * 16u;             // byte col within row
            uint32_t vl  = vbuf_lds[sel] + n * 128u + co;
            uint32_t vvo = n * 8192u + co;
            asm volatile("global_load_async_to_lds_b128 %0, %1, %2 offset:0"
                         :: "v"(vl), "v"(vvo), "s"(v_gb) : "memory");
        }
    };

    // ---- Q tile A-registers (pre-scaled by 1/sqrt(dk) in projection) ------
    AB16 qA[4];
    {
        const bf16* qrow = qbase + (size_t)(m0 + lane16) * D_HEAD + hi * 8;
#pragma unroll
        for (int c = 0; c < 4; ++c) {
            qA[c].q[0] = *(const uint4*)(qrow + c * 32);
            qA[c].q[1] = *(const uint4*)(qrow + c * 32 + 16);
        }
    }

    const v8f zero8 = {0.f, 0.f, 0.f, 0.f, 0.f, 0.f, 0.f, 0.f};
    v8f oacc[8];
#pragma unroll
    for (int t = 0; t < 8; ++t) oacc[t] = zero8;

    float mrow[8], lrow[8];
#pragma unroll
    for (int r = 0; r < 8; ++r) { mrow[r] = -INFINITY; lrow[r] = 0.f; }

    stage_chunk(0, 0);
    int cur = 0;

    for (int j = 0; j < S_LEN; j += 64) {
        if (j + 64 < S_LEN) {
            stage_chunk(cur ^ 1, j + 64);
            asm volatile("s_wait_asynccnt 0x8" ::: "memory");
        } else {
            asm volatile("s_wait_asynccnt 0x0" ::: "memory");
        }
        __syncthreads();  // all waves' slices of `cur` visible

        const bf16* kc_ = &kbuf[cur][0];
        const bf16* vc_ = &vbuf[cur][0];

        // ---- scores: 16x64 = q(16x128) @ k_chunk^T (from LDS) ----
        v8f sc[4];
#pragma unroll
        for (int ns = 0; ns < 4; ++ns) {
            sc[ns] = zero8;
            const bf16* krow = kc_ + (ns * 16 + lane16) * D_HEAD + hi * 8;
#pragma unroll
            for (int c = 0; c < 4; ++c) {
                AB16 bm;
                bm.q[0] = *(const uint4*)(krow + c * 32);
                bm.q[1] = *(const uint4*)(krow + c * 32 + 16);
                sc[ns] = __builtin_amdgcn_wmma_f32_16x16x32_bf16(
                    false, qA[c].v, false, bm.v, (short)0, sc[ns], false, false);
            }
        }

        // ---- online softmax (reductions across 16-lane half-groups) ----
#pragma unroll
        for (int r = 0; r < 8; ++r) {
            float v0 = fmaxf(fmaxf(sc[0][r], sc[1][r]), fmaxf(sc[2][r], sc[3][r]));
#pragma unroll
            for (int off = 1; off < 16; off <<= 1)
                v0 = fmaxf(v0, __shfl_xor(v0, off, 16));
            const float mnew  = fmaxf(mrow[r], v0);
            const float alpha = __expf(mrow[r] - mnew);  // 0 on first chunk
            float s0 = 0.f;
#pragma unroll
            for (int ns = 0; ns < 4; ++ns) {
                float p = __expf(sc[ns][r] - mnew);
                sc[ns][r] = p;
                s0 += p;
            }
#pragma unroll
            for (int off = 1; off < 16; off <<= 1)
                s0 += __shfl_xor(s0, off, 16);
            lrow[r] = lrow[r] * alpha + s0;
            mrow[r] = mnew;
#pragma unroll
            for (int nt = 0; nt < 8; ++nt) oacc[nt][r] *= alpha;
        }

        // ---- P: D layout -> LDS -> A layout (same-wave, DS in-order) ----
        bf16* pdst = &lds_p[wave * 1024];
#pragma unroll
        for (int ns = 0; ns < 4; ++ns)
#pragma unroll
            for (int r = 0; r < 8; ++r)
                pdst[(r + hi * 8) * 64 + ns * 16 + lane16] = (bf16)sc[ns][r];

        AB16 pA[2];
        {
            const bf16* psrc = &lds_p[wave * 1024 + lane16 * 64 + hi * 8];
#pragma unroll
            for (int c2 = 0; c2 < 2; ++c2) {
                pA[c2].q[0] = *(const uint4*)(psrc + c2 * 32);
                pA[c2].q[1] = *(const uint4*)(psrc + c2 * 32 + 16);
            }
        }

        // ---- O += P(16x64) @ V_chunk(64x128), V^T rows from LDS ----
#pragma unroll
        for (int nt = 0; nt < 8; ++nt) {
            const bf16* vrow = vc_ + (nt * 16 + lane16) * 64 + hi * 8;
#pragma unroll
            for (int c2 = 0; c2 < 2; ++c2) {
                AB16 bm;
                bm.q[0] = *(const uint4*)(vrow + c2 * 32);
                bm.q[1] = *(const uint4*)(vrow + c2 * 32 + 16);
                oacc[nt] = __builtin_amdgcn_wmma_f32_16x16x32_bf16(
                    false, pA[c2].v, false, bm.v, (short)0, oacc[nt], false, false);
            }
        }

        __syncthreads();  // all waves done reading `cur` before it is re-filled
        cur ^= 1;
    }

    // ---- normalize + store fp32 output ----
#pragma unroll
    for (int r = 0; r < 8; ++r) {
        const float inv = 1.0f / lrow[r];
        const int mm = m0 + r + hi * 8;
        float* orow = out + ((size_t)b * S_LEN + mm) * D_HEAD + lane16;
#pragma unroll
        for (int nt = 0; nt < 8; ++nt) orow[nt * 16] = oacc[nt][r] * inv;
    }
}

// ---------------------------------------------------------------------------
extern "C" void kernel_launch(void* const* d_in, const int* in_sizes, int n_in,
                              void* d_out, int out_size, void* d_ws, size_t ws_size,
                              hipStream_t stream) {
    const float* Q  = (const float*)d_in[0];
    const float* K  = (const float*)d_in[1];
    const float* V  = (const float*)d_in[2];
    const float* Wq = (const float*)d_in[3];
    const float* bq = (const float*)d_in[4];
    const float* Wk = (const float*)d_in[5];
    const float* bk = (const float*)d_in[6];
    const float* Wv = (const float*)d_in[7];
    const float* bv = (const float*)d_in[8];

    const size_t elems = (size_t)4 * S_LEN * D_HEAD;  // 2,097,152
    bf16* qb  = (bf16*)d_ws;
    bf16* kbf = qb + elems;
    bf16* vtb = kbf + elems;

    const float sm_scale = 0.0883883476483184f;  // 1/sqrt(128), folded into q

    dim3 pgrid(16384 / 128);
    proj_kernel<<<pgrid, 256, 0, stream>>>(Q, Wq, bq, qb, 0, sm_scale);
    proj_kernel<<<pgrid, 256, 0, stream>>>(K, Wk, bk, kbf, 0, 1.0f);
    proj_kernel<<<pgrid, 256, 0, stream>>>(V, Wv, bv, vtb, 1, 1.0f);  // V^T

    attn_kernel<<<dim3(S_LEN / 128, 4), 256, 0, stream>>>(qb, kbf, vtb,
                                                          (float*)d_out);
}
